// Attention_62921270887067
// MI455X (gfx1250) — matrix operand order repaired and checked by reading, not compiled, not run on hardware
//
#include <hip/hip_runtime.h>

// ---------------------------------------------------------------------------
// Types for CDNA5 WMMA (gfx1250, wave32)
// ---------------------------------------------------------------------------
typedef __bf16 bf16;
typedef __attribute__((ext_vector_type(8)))  __bf16 v8bf;
typedef __attribute__((ext_vector_type(16))) __bf16 v16bf;
typedef __attribute__((ext_vector_type(8)))  float  v8f;
typedef int v4i __attribute__((__vector_size__(4 * sizeof(int))));

#define WMMA_BF16(a, b, c) \
  __builtin_amdgcn_wmma_f32_16x16x32_bf16(false, (a), false, (b), (short)0, (c), false, false)

#define AS1 __attribute__((address_space(1)))
#define AS3 __attribute__((address_space(3)))

// CDNA5 async global->LDS copy path (ASYNCcnt), with synchronous fallback.
#if defined(__AMDGCN__) && defined(__has_builtin)
#if __has_builtin(__builtin_amdgcn_global_load_async_to_lds_b128)
#define ASYNC_COPY 1
#endif
#endif

__device__ __forceinline__ void copy16(const bf16* g, bf16* l) {
#ifdef ASYNC_COPY
  __builtin_amdgcn_global_load_async_to_lds_b128((AS1 v4i*)(g), (AS3 v4i*)(l),
                                                 0, 0);
#else
  *(v8bf*)l = *(const v8bf*)g;
#endif
}

__device__ __forceinline__ void async_join() {
#ifdef ASYNC_COPY
#if __has_builtin(__builtin_amdgcn_s_wait_asynccnt)
  __builtin_amdgcn_s_wait_asynccnt(0);
#else
  asm volatile("s_wait_asynccnt 0" ::: "memory");
#endif
#endif
}

__device__ __forceinline__ bf16 f2bf(float f) {
  union { float f; unsigned u; } in; in.f = f;
  unsigned r = (in.u + 0x7FFFu + ((in.u >> 16) & 1u)) >> 16;  // RNE
  union { unsigned short s; bf16 b; } out; out.s = (unsigned short)r;
  return out.b;
}

// A-fragment (16x32, bf16): lane L holds row M = L&15.
// Lanes 0-15: K = {0..7, 16..23}; lanes 16-31: K = {8..15, 24..31}. (ISA 7.12.2)
__device__ __forceinline__ v16bf load_fragA(const bf16* base, int ld) {
  const int lane = threadIdx.x & 31;
  const bf16* p = base + (lane & 15) * ld + ((lane & 16) ? 8 : 0);
  v8bf lo = *(const v8bf*)(p);
  v8bf hi = *(const v8bf*)(p + 16);
  return __builtin_shufflevector(lo, hi, 0, 1, 2, 3, 4, 5, 6, 7,
                                         8, 9, 10, 11, 12, 13, 14, 15);
}

// B-fragment (32x16, bf16) from a tile stored transposed ([N][K] row-major):
// lane L holds column N = L&15; lanes 0-15: K={0..7,16..23}, lanes 16-31: K={8..15,24..31}.
__device__ __forceinline__ v16bf load_fragB(const bf16* baseT, int ld) {
  const int lane = threadIdx.x & 31;
  const bf16* p = baseT + (lane & 15) * ld + ((lane & 16) ? 8 : 0);
  v8bf lo = *(const v8bf*)(p);
  v8bf hi = *(const v8bf*)(p + 16);
  return __builtin_shufflevector(lo, hi, 0, 1, 2, 3, 4, 5, 6, 7,
                                         8, 9, 10, 11, 12, 13, 14, 15);
}

// ---------------------------------------------------------------------------
// Problem constants
// ---------------------------------------------------------------------------
#define BATCH 4
#define SEQ   2048
#define DIM   1024
#define HEADS 16
#define DHEAD 64
#define ROWS  (BATCH * SEQ)     // 8192
#define NQKV  (3 * DIM)         // 3072

// ---------------------------------------------------------------------------
// Pre-pass: fp32 -> bf16 conversions / weight transposes
// ---------------------------------------------------------------------------
__global__ void convert_x_kernel(const float* __restrict__ x,
                                 bf16* __restrict__ xb, int n) {
  for (int i = blockIdx.x * blockDim.x + threadIdx.x; i < n;
       i += gridDim.x * blockDim.x)
    xb[i] = f2bf(x[i]);
}

// w: [rows][cols] fp32 -> wT: [cols][rows] bf16
__global__ void transpose_w_kernel(const float* __restrict__ w,
                                   bf16* __restrict__ wT, int rows, int cols) {
  int n = rows * cols;
  for (int i = blockIdx.x * blockDim.x + threadIdx.x; i < n;
       i += gridDim.x * blockDim.x) {
    int r = i / cols, c = i - r * cols;
    wT[(size_t)c * rows + r] = f2bf(w[i]);
  }
}

// ---------------------------------------------------------------------------
// GEMM 1: qkv = xb[8192,1024] @ w_qkv  (B given transposed: wT[3072][1024])
// Fused scatter epilogue -> Q[b,h,n,d], K[b,h,n,d], Vt[b,h,d,n] (all bf16)
// Block: 128x128 tile, 8 waves each 32x64. Double-buffered 64-deep K tiles,
// staged with async global->LDS copies.
// ---------------------------------------------------------------------------
__global__ __launch_bounds__(256) void qkv_gemm_kernel(
    const bf16* __restrict__ xb, const bf16* __restrict__ wT,
    bf16* __restrict__ qbuf, bf16* __restrict__ kbuf,
    bf16* __restrict__ vtbuf) {
  __shared__ alignas(32) bf16 As[2][128][64];
  __shared__ alignas(32) bf16 Bs[2][128][64];
  const int tid = threadIdx.x;
  const int wave = tid >> 5, lane = tid & 31;
  const int ln = lane & 15, hi8 = (lane & 16) ? 8 : 0;
  const int m0 = blockIdx.y * 128, n0 = blockIdx.x * 128;
  const int wm = (wave >> 1) * 32, wn = (wave & 1) * 64;

  v8f acc[2][4];
#pragma unroll
  for (int i = 0; i < 2; ++i)
#pragma unroll
    for (int j = 0; j < 4; ++j)
#pragma unroll
      for (int v = 0; v < 8; ++v) acc[i][j][v] = 0.f;

  // tile fetch: 128x64 bf16 per matrix = 1024 16B chunks, 4 per thread each
#define QKV_FETCH(k0, buf)                                                    \
  do {                                                                        \
    _Pragma("unroll") for (int i = 0; i < 4; ++i) {                           \
      int c = tid + i * 256;                                                  \
      int r = c >> 3, cc = (c & 7) * 8;                                       \
      copy16(&xb[(size_t)(m0 + r) * DIM + (k0) + cc], &As[buf][r][cc]);       \
      copy16(&wT[(size_t)(n0 + r) * DIM + (k0) + cc], &Bs[buf][r][cc]);       \
    }                                                                         \
  } while (0)

  QKV_FETCH(0, 0);
  async_join();
  __syncthreads();

  int cur = 0;
  for (int k0 = 0; k0 < DIM; k0 += 64) {
    if (k0 + 64 < DIM) QKV_FETCH(k0 + 64, cur ^ 1);
#pragma unroll
    for (int ks = 0; ks < 64; ks += 32) {
      v16bf a[2], b[4];
#pragma unroll
      for (int i = 0; i < 2; ++i)
        a[i] = load_fragA(&As[cur][wm + i * 16][ks], 64);
#pragma unroll
      for (int j = 0; j < 4; ++j)
        b[j] = load_fragB(&Bs[cur][wn + j * 16][ks], 64);
#pragma unroll
      for (int i = 0; i < 2; ++i)
#pragma unroll
        for (int j = 0; j < 4; ++j) acc[i][j] = WMMA_BF16(a[i], b[j], acc[i][j]);
    }
    async_join();
    __syncthreads();
    cur ^= 1;
  }
#undef QKV_FETCH

  // C layout: lane ln -> column; vgpr v -> row v (+8 for lanes 16-31).
#pragma unroll
  for (int i = 0; i < 2; ++i)
#pragma unroll
    for (int j = 0; j < 4; ++j) {
      int gn = n0 + wn + j * 16 + ln;          // [0, 3072)
      int which = gn >> 10;                    // 0=q 1=k 2=v
      int h = (gn >> 6) & 15, dd = gn & 63;
#pragma unroll
      for (int v = 0; v < 8; ++v) {
        int gm = m0 + wm + i * 16 + hi8 + v;   // [0, 8192)
        int bb = gm >> 11, nn = gm & 2047;
        int bh = bb * HEADS + h;
        bf16 val = f2bf(acc[i][j][v]);
        if (which == 0)
          qbuf[((size_t)bh * SEQ + nn) * DHEAD + dd] = val;
        else if (which == 1)
          kbuf[((size_t)bh * SEQ + nn) * DHEAD + dd] = val;
        else
          vtbuf[((size_t)bh * DHEAD + dd) * SEQ + nn] = val;
      }
    }
}

// ---------------------------------------------------------------------------
// Attention: per (b,h, 128-query tile), flash-style over 64-key chunks.
// Q resident in LDS; K / Vt chunks double-buffered via async copies.
// Each wave owns 16 query rows. Writes attout[b*n][h*d] bf16.
// ---------------------------------------------------------------------------
__global__ __launch_bounds__(256) void attn_kernel(
    const bf16* __restrict__ qbuf, const bf16* __restrict__ kbuf,
    const bf16* __restrict__ vtbuf, bf16* __restrict__ attout) {
  __shared__ alignas(32) bf16 Qs[128][64];        // 16 KB
  __shared__ alignas(32) bf16 Ks[2][64][64];      // 16 KB
  __shared__ alignas(32) bf16 Vs[2][64][64];      // 16 KB  (Vt chunk: [d][key])
  __shared__ alignas(32) bf16 Ps[8][16][64];      // 16 KB  per-wave P tiles

  const int tid = threadIdx.x;
  const int wave = tid >> 5, lane = tid & 31;
  const int ln = lane & 15, hi8 = (lane & 16) ? 8 : 0;
  const int qt = blockIdx.x, h = blockIdx.y, b = blockIdx.z;
  const int bh = b * HEADS + h;
  const int q0 = qt * 128;
  const float scale = 0.125f;  // 64^-0.5

#define KV_FETCH(kc, buf)                                                     \
  do {                                                                        \
    _Pragma("unroll") for (int i = 0; i < 2; ++i) {                           \
      int c = tid + i * 256;                                                  \
      int r = c >> 3, cc = (c & 7) * 8;                                       \
      copy16(&kbuf[((size_t)bh * SEQ + (kc) + r) * DHEAD + cc],               \
             &Ks[buf][r][cc]);                                                \
      copy16(&vtbuf[((size_t)bh * DHEAD + r) * SEQ + (kc) + cc],              \
             &Vs[buf][r][cc]);                                                \
    }                                                                         \
  } while (0)

  // Q tile (128x64) + first K/V chunk, one async join for all of it.
#pragma unroll
  for (int i = 0; i < 4; ++i) {
    int c = tid + i * 256;
    int r = c >> 3, cc = (c & 7) * 8;
    copy16(&qbuf[((size_t)bh * SEQ + q0 + r) * DHEAD + cc], &Qs[r][cc]);
  }
  KV_FETCH(0, 0);
  async_join();
  __syncthreads();

  v8f Oacc[4];
  float m8[8], l8[8];
#pragma unroll
  for (int j = 0; j < 4; ++j)
#pragma unroll
    for (int v = 0; v < 8; ++v) Oacc[j][v] = 0.f;
#pragma unroll
  for (int v = 0; v < 8; ++v) { m8[v] = -1e30f; l8[v] = 0.f; }

  int cur = 0;
  for (int kc = 0; kc < SEQ; kc += 64) {
    if (kc + 64 < SEQ) KV_FETCH(kc + 64, cur ^ 1);

    // S = Q K^T  (16 rows x 64 keys per wave)
    v8f S[4];
#pragma unroll
    for (int j = 0; j < 4; ++j)
#pragma unroll
      for (int v = 0; v < 8; ++v) S[j][v] = 0.f;
#pragma unroll
    for (int ks = 0; ks < 64; ks += 32) {
      v16bf a = load_fragA(&Qs[wave * 16][ks], 64);
#pragma unroll
      for (int j = 0; j < 4; ++j) {
        v16bf bf = load_fragB(&Ks[cur][j * 16][ks], 64);
        S[j] = WMMA_BF16(a, bf, S[j]);
      }
    }

    // Online softmax. Row M lives in vgpr v across the 16-lane half-group.
#pragma unroll
    for (int v = 0; v < 8; ++v) {
#pragma unroll
      for (int j = 0; j < 4; ++j) S[j][v] *= scale;
      float mx = fmaxf(fmaxf(S[0][v], S[1][v]), fmaxf(S[2][v], S[3][v]));
      mx = fmaxf(mx, __shfl_xor(mx, 1, 32));
      mx = fmaxf(mx, __shfl_xor(mx, 2, 32));
      mx = fmaxf(mx, __shfl_xor(mx, 4, 32));
      mx = fmaxf(mx, __shfl_xor(mx, 8, 32));
      float mnew = fmaxf(m8[v], mx);
      float alpha = __expf(m8[v] - mnew);
      m8[v] = mnew;
      l8[v] *= alpha;
#pragma unroll
      for (int j = 0; j < 4; ++j) Oacc[j][v] *= alpha;
      float rs = 0.f;
#pragma unroll
      for (int j = 0; j < 4; ++j) {
        float p = __expf(S[j][v] - mnew);
        S[j][v] = p;
        rs += p;
      }
      rs += __shfl_xor(rs, 1, 32);
      rs += __shfl_xor(rs, 2, 32);
      rs += __shfl_xor(rs, 4, 32);
      rs += __shfl_xor(rs, 8, 32);
      l8[v] += rs;
    }

    // P (C-layout) -> wave-private LDS in A-layout source form
#pragma unroll
    for (int j = 0; j < 4; ++j)
#pragma unroll
      for (int v = 0; v < 8; ++v)
        Ps[wave][hi8 + v][j * 16 + ln] = f2bf(S[j][v]);
    // DS ops are in-order per wave; drain dscnt before fragment re-load.
    asm volatile("s_wait_dscnt 0" ::: "memory");

    // O += P @ V   (B-frags come from transposed V chunk, contiguous rows)
#pragma unroll
    for (int ks = 0; ks < 64; ks += 32) {
      v16bf pa = load_fragA(&Ps[wave][0][ks], 64);
#pragma unroll
      for (int j = 0; j < 4; ++j) {
        v16bf vb = load_fragB(&Vs[cur][j * 16][ks], 64);
        Oacc[j] = WMMA_BF16(pa, vb, Oacc[j]);
      }
    }

    async_join();
    __syncthreads();
    cur ^= 1;
  }
#undef KV_FETCH

  // attout layout: [b*n][h*d] row-major (feeds the output projection GEMM)
#pragma unroll
  for (int j = 0; j < 4; ++j)
#pragma unroll
    for (int v = 0; v < 8; ++v) {
      int gm = q0 + wave * 16 + hi8 + v;        // n index
      int gd = h * DHEAD + j * 16 + ln;         // [0, 1024)
      float o = Oacc[j][v] / l8[v];
      attout[((size_t)b * SEQ + gm) * DIM + gd] = f2bf(o);
    }
}

// ---------------------------------------------------------------------------
// GEMM 3: out = attout[8192,1024] @ w_out  (woutT[1024][1024] bf16), fp32 out
// ---------------------------------------------------------------------------
__global__ __launch_bounds__(256) void out_gemm_kernel(
    const bf16* __restrict__ a, const bf16* __restrict__ wT,
    float* __restrict__ out) {
  __shared__ alignas(32) bf16 As[2][128][64];
  __shared__ alignas(32) bf16 Bs[2][128][64];
  const int tid = threadIdx.x;
  const int wave = tid >> 5, lane = tid & 31;
  const int ln = lane & 15, hi8 = (lane & 16) ? 8 : 0;
  const int m0 = blockIdx.y * 128, n0 = blockIdx.x * 128;
  const int wm = (wave >> 1) * 32, wn = (wave & 1) * 64;

  v8f acc[2][4];
#pragma unroll
  for (int i = 0; i < 2; ++i)
#pragma unroll
    for (int j = 0; j < 4; ++j)
#pragma unroll
      for (int v = 0; v < 8; ++v) acc[i][j][v] = 0.f;

#define OUT_FETCH(k0, buf)                                                    \
  do {                                                                        \
    _Pragma("unroll") for (int i = 0; i < 4; ++i) {                           \
      int c = tid + i * 256;                                                  \
      int r = c >> 3, cc = (c & 7) * 8;                                       \
      copy16(&a[(size_t)(m0 + r) * DIM + (k0) + cc], &As[buf][r][cc]);        \
      copy16(&wT[(size_t)(n0 + r) * DIM + (k0) + cc], &Bs[buf][r][cc]);       \
    }                                                                         \
  } while (0)

  OUT_FETCH(0, 0);
  async_join();
  __syncthreads();

  int cur = 0;
  for (int k0 = 0; k0 < DIM; k0 += 64) {
    if (k0 + 64 < DIM) OUT_FETCH(k0 + 64, cur ^ 1);
#pragma unroll
    for (int ks = 0; ks < 64; ks += 32) {
      v16bf af[2], bf[4];
#pragma unroll
      for (int i = 0; i < 2; ++i)
        af[i] = load_fragA(&As[cur][wm + i * 16][ks], 64);
#pragma unroll
      for (int j = 0; j < 4; ++j)
        bf[j] = load_fragB(&Bs[cur][wn + j * 16][ks], 64);
#pragma unroll
      for (int i = 0; i < 2; ++i)
#pragma unroll
        for (int j = 0; j < 4; ++j)
          acc[i][j] = WMMA_BF16(af[i], bf[j], acc[i][j]);
    }
    async_join();
    __syncthreads();
    cur ^= 1;
  }
#undef OUT_FETCH

#pragma unroll
  for (int i = 0; i < 2; ++i)
#pragma unroll
    for (int j = 0; j < 4; ++j) {
      int gn = n0 + wn + j * 16 + ln;
#pragma unroll
      for (int v = 0; v < 8; ++v) {
        int gm = m0 + wm + i * 16 + hi8 + v;
        out[(size_t)gm * DIM + gn] = acc[i][j][v];
      }
    }
}

// ---------------------------------------------------------------------------
// Host launcher
// ---------------------------------------------------------------------------
extern "C" void kernel_launch(void* const* d_in, const int* in_sizes, int n_in,
                              void* d_out, int out_size, void* d_ws,
                              size_t ws_size, hipStream_t stream) {
  const float* x     = (const float*)d_in[0];  // [4,2048,1024]
  const float* w_qkv = (const float*)d_in[1];  // [1024,3072]
  const float* w_out = (const float*)d_in[2];  // [1024,1024]
  float* out = (float*)d_out;                  // [4,2048,1024] fp32

  char* ws = (char*)d_ws;
  size_t off = 0;
  bf16* xb     = (bf16*)(ws + off); off += (size_t)ROWS * DIM * 2;       // 16 MB
  bf16* wqkvT  = (bf16*)(ws + off); off += (size_t)NQKV * DIM * 2;       //  6 MB
  bf16* woutT  = (bf16*)(ws + off); off += (size_t)DIM * DIM * 2;        //  2 MB
  bf16* qbuf   = (bf16*)(ws + off); off += (size_t)ROWS * DIM * 2;       // 16 MB
  bf16* kbuf   = (bf16*)(ws + off); off += (size_t)ROWS * DIM * 2;       // 16 MB
  bf16* vtbuf  = (bf16*)(ws + off); off += (size_t)ROWS * DIM * 2;       // 16 MB
  bf16* attout = (bf16*)(ws + off); off += (size_t)ROWS * DIM * 2;       // 16 MB
  (void)off; (void)ws_size; (void)in_sizes; (void)n_in; (void)out_size;

  convert_x_kernel<<<4096, 256, 0, stream>>>(x, xb, ROWS * DIM);
  transpose_w_kernel<<<2048, 256, 0, stream>>>(w_qkv, wqkvT, DIM, NQKV);
  transpose_w_kernel<<<1024, 256, 0, stream>>>(w_out, woutT, DIM, DIM);

  qkv_gemm_kernel<<<dim3(NQKV / 128, ROWS / 128), 256, 0, stream>>>(
      xb, wqkvT, qbuf, kbuf, vtbuf);

  attn_kernel<<<dim3(SEQ / 128, HEADS, BATCH), 256, 0, stream>>>(
      qbuf, kbuf, vtbuf, attout);

  out_gemm_kernel<<<dim3(DIM / 128, ROWS / 128), 256, 0, stream>>>(
      attout, woutT, out);
}
